// LMFloss_77129022702017
// MI455X (gfx1250) — matrix operand order repaired and checked by reading, not compiled, not run on hardware
//
#include <hip/hip_runtime.h>
#include <math.h>

#define B_ROWS 16384
#define C_CLS  2000
#define F_DIM  1024
#define INV_B  (1.0f/16384.0f)
#define MB     64          // rows per block
#define KPAD   1032        // padded row length in bf16 elems (+16B pad -> bank stagger)
#define NT     125         // 2000 / 16 column tiles

// workspace layout (bytes):
//   [0]   fisher accumulator (float)
//   [4]   aux accumulator (float)
//   [16]  true_logit[B] floats                  (ends at 16 + 65536 = 65552, 16B aligned)
//   [65552]          Whi bf16 plane [C][F]      (4,096,000 B)
//   [4161552]        Wlo bf16 plane [C][F]      (4,096,000 B)
#define WS_TLOGIT_F   4                       // float index of tlogit[0]
#define WS_WHI_BYTES  65552u
#define WS_WLO_BYTES  (65552u + 4096000u)
#define WS_NEEDED     (65552u + 8192000u)

typedef __attribute__((ext_vector_type(16))) __bf16 v16bf;
typedef __attribute__((ext_vector_type(8)))  float  v8f;

__device__ __forceinline__ unsigned short f2bf(float f) {
    unsigned u = __float_as_uint(f);
    unsigned r = 0x7FFFu + ((u >> 16) & 1u);
    return (unsigned short)((u + r) >> 16);
}
__device__ __forceinline__ float bf2f(unsigned short h) {
    return __uint_as_float(((unsigned)h) << 16);
}

union V16U {
    v16bf v;
    unsigned short u[16];
    uint4 q[2];
};

__global__ void k_init(float* wsf) {
    if (threadIdx.x < 2) wsf[threadIdx.x] = 0.0f;
}

// ---------------- W -> (hi, lo) bf16 planes, once (8 MB, bandwidth trivial)
__global__ __launch_bounds__(256)
void k_convW(const float* __restrict__ W,
             unsigned short* __restrict__ whi,
             unsigned short* __restrict__ wlo) {
    int idx = blockIdx.x * 256 + threadIdx.x;        // one float4 per thread
    if (idx >= (C_CLS * F_DIM) / 4) return;
    float4 x = ((const float4*)W)[idx];
    unsigned short h0 = f2bf(x.x), h1 = f2bf(x.y), h2 = f2bf(x.z), h3 = f2bf(x.w);
    unsigned short l0 = f2bf(x.x - bf2f(h0)), l1 = f2bf(x.y - bf2f(h1));
    unsigned short l2 = f2bf(x.z - bf2f(h2)), l3 = f2bf(x.w - bf2f(h3));
    ((uint2*)whi)[idx] = make_uint2((unsigned)h0 | ((unsigned)h1 << 16),
                                    (unsigned)h2 | ((unsigned)h3 << 16));
    ((uint2*)wlo)[idx] = make_uint2((unsigned)l0 | ((unsigned)l1 << 16),
                                    (unsigned)l2 | ((unsigned)l3 << 16));
}

// ---------------- pass 1: fisher center loss + true-class logit (bandwidth bound)
__global__ __launch_bounds__(256)
void k_fisher_tlogit(const float* __restrict__ feat,
                     const long long* __restrict__ labels,
                     const float* __restrict__ centers,
                     const float* __restrict__ W,
                     const float* __restrict__ bias,
                     float* __restrict__ wsf) {
    const int wave = threadIdx.x >> 5, lane = threadIdx.x & 31;
    const int row  = blockIdx.x * 8 + wave;
    const long long lab = labels[row];
    const float4* xf = (const float4*)(feat    + (size_t)row * F_DIM);
    const float4* cf = (const float4*)(centers + (size_t)lab * F_DIM);
    const float4* wf = (const float4*)(W       + (size_t)lab * F_DIM);
    float fs = 0.f, tl = 0.f;
    for (int i = lane; i < F_DIM / 4; i += 32) {
        float4 x = xf[i], c = cf[i], w = wf[i];
        float dx = x.x - c.x, dy = x.y - c.y, dz = x.z - c.z, dw = x.w - c.w;
        fs += dx*dx + dy*dy + dz*dz + dw*dw;
        tl += x.x*w.x + x.y*w.y + x.z*w.z + x.w*w.w;
    }
    for (int off = 16; off; off >>= 1) {
        fs += __shfl_down(fs, off, 32);
        tl += __shfl_down(tl, off, 32);
    }
    if (lane == 0) {
        atomicAdd(&wsf[0], fs * INV_B);
        wsf[WS_TLOGIT_F + row] = tl + bias[lab];
    }
}

// ---------------- pass 2: logits = feat @ W^T + b via split-bf16 WMMA, fused online logsumexp
struct SmemA { unsigned short ahi[MB * KPAD]; unsigned short alo[MB * KPAD]; };
union  Smem  { SmemA a; float2 red[MB * 128]; };

template <bool PRE>
__global__ __launch_bounds__(256)
void k_logits_lse(const float* __restrict__ feat,
                  const float* __restrict__ W,
                  const unsigned short* __restrict__ whi,
                  const unsigned short* __restrict__ wlo,
                  const float* __restrict__ bias,
                  float* __restrict__ wsf) {
    __shared__ Smem sm;
    const int tid  = threadIdx.x;
    const int wave = tid >> 5, lane = tid & 31;
    const int row0 = blockIdx.x * MB;
    const int nhalf = lane >> 4;      // 0: lanes 0-15, 1: lanes 16-31
    const int nl    = lane & 15;

    // ---- stage A block (64 rows x 1024) as hi/lo bf16 planes in LDS (once per block)
    for (int idx = tid; idx < MB * (F_DIM / 4); idx += 256) {
        int r  = idx >> 8;            // F_DIM/4 == 256
        int c4 = idx & 255;
        float4 x = ((const float4*)(feat + (size_t)(row0 + r) * F_DIM))[c4];
        unsigned short h0 = f2bf(x.x), h1 = f2bf(x.y), h2 = f2bf(x.z), h3 = f2bf(x.w);
        unsigned short l0 = f2bf(x.x - bf2f(h0)), l1 = f2bf(x.y - bf2f(h1));
        unsigned short l2 = f2bf(x.z - bf2f(h2)), l3 = f2bf(x.w - bf2f(h3));
        *(uint2*)&sm.a.ahi[r * KPAD + c4 * 4] =
            make_uint2((unsigned)h0 | ((unsigned)h1 << 16), (unsigned)h2 | ((unsigned)h3 << 16));
        *(uint2*)&sm.a.alo[r * KPAD + c4 * 4] =
            make_uint2((unsigned)l0 | ((unsigned)l1 << 16), (unsigned)l2 | ((unsigned)l3 << 16));
    }
    __syncthreads();

    // ---- online logsumexp state: each lane owns disjoint column subset of its rows
    v8f smax[4], ssum[4];
    #pragma unroll
    for (int m = 0; m < 4; ++m)
        #pragma unroll
        for (int j = 0; j < 8; ++j) { smax[m][j] = -__builtin_inff(); ssum[m][j] = 0.f; }

    for (int t = wave; t < NT; t += 8) {
        const int c0 = t * 16;
        const float bv = bias[c0 + nl];
        v8f acc[4];
        #pragma unroll
        for (int m = 0; m < 4; ++m)
            #pragma unroll
            for (int j = 0; j < 8; ++j) acc[m][j] = bv;   // fold bias into C

        // B-fragment source row: lane nl = column c0+nl; lanes 0-15: K k0..k0+15, 16-31: +16
        const size_t wbase = (size_t)(c0 + nl) * F_DIM + nhalf * 16;
        #pragma unroll 1
        for (int k0 = 0; k0 < F_DIM; k0 += 32) {
            V16U bh, bl;
            if constexpr (PRE) {
                const unsigned short* ph = whi + wbase + k0;
                const unsigned short* pl = wlo + wbase + k0;
                bh.q[0] = *(const uint4*)ph;  bh.q[1] = *(const uint4*)(ph + 8);
                bl.q[0] = *(const uint4*)pl;  bl.q[1] = *(const uint4*)(pl + 8);
            } else {
                const float4* wp = (const float4*)(W + wbase + k0);
                #pragma unroll
                for (int q = 0; q < 4; ++q) {
                    float4 w4 = wp[q];
                    unsigned short h0 = f2bf(w4.x), h1 = f2bf(w4.y),
                                   h2 = f2bf(w4.z), h3 = f2bf(w4.w);
                    bh.u[q*4+0] = h0; bh.u[q*4+1] = h1; bh.u[q*4+2] = h2; bh.u[q*4+3] = h3;
                    bl.u[q*4+0] = f2bf(w4.x - bf2f(h0));
                    bl.u[q*4+1] = f2bf(w4.y - bf2f(h1));
                    bl.u[q*4+2] = f2bf(w4.z - bf2f(h2));
                    bl.u[q*4+3] = f2bf(w4.w - bf2f(h3));
                }
            }
            const int asel = nhalf * 16;  // lanes 0-15: K {0..15}; 16-31: K {16..31}
            #pragma unroll
            for (int m = 0; m < 4; ++m) {
                const unsigned short* ah_p = &sm.a.ahi[(m * 16 + nl) * KPAD + k0 + asel];
                const unsigned short* al_p = &sm.a.alo[(m * 16 + nl) * KPAD + k0 + asel];
                V16U ah, al;
                ah.q[0] = *(const uint4*)ah_p;
                ah.q[1] = *(const uint4*)(ah_p + 8);
                al.q[0] = *(const uint4*)al_p;
                al.q[1] = *(const uint4*)(al_p + 8);
                // split product: hi*hi + hi*lo + lo*hi  (~16-bit mantissa fidelity)
                acc[m] = __builtin_amdgcn_wmma_f32_16x16x32_bf16(false, ah.v, false, bh.v,
                                                                 (short)0, acc[m], false, false);
                acc[m] = __builtin_amdgcn_wmma_f32_16x16x32_bf16(false, ah.v, false, bl.v,
                                                                 (short)0, acc[m], false, false);
                acc[m] = __builtin_amdgcn_wmma_f32_16x16x32_bf16(false, al.v, false, bh.v,
                                                                 (short)0, acc[m], false, false);
            }
        }
        // online (max, sum) update with this tile's 16x64 logits
        #pragma unroll
        for (int m = 0; m < 4; ++m)
            #pragma unroll
            for (int j = 0; j < 8; ++j) {
                float x  = acc[m][j];
                float om = smax[m][j];
                float nm = fmaxf(om, x);
                ssum[m][j] = ssum[m][j] * __expf(om - nm) + __expf(x - nm);
                smax[m][j] = nm;
            }
    }
    __syncthreads();   // all waves done reading A planes; reuse LDS for reduction

    // deposit (m,s): C/D layout -> row = 16m + j + 8*nhalf, column slot = wave*16 + nl
    #pragma unroll
    for (int m = 0; m < 4; ++m)
        #pragma unroll
        for (int j = 0; j < 8; ++j) {
            int r = m * 16 + j + nhalf * 8;
            sm.red[r * 128 + wave * 16 + nl] = make_float2(smax[m][j], ssum[m][j]);
        }
    __syncthreads();

    if (tid < MB) {
        float M = -__builtin_inff(), S = 0.f;
        const float2* p = &sm.red[tid * 128];
        for (int j = 0; j < 128; ++j) {
            float2 e = p[j];
            float nm = fmaxf(M, e.x);
            S = S * __expf(M - nm) + e.y * __expf(e.x - nm);
            M = nm;
        }
        float logZ = M + __logf(S);
        int grow = row0 + tid;
        atomicAdd(&wsf[1], (logZ - wsf[WS_TLOGIT_F + grow]) * INV_B);
    }
}

__global__ void k_final(const float* __restrict__ wsf, float* __restrict__ out) {
    if (threadIdx.x == 0) out[0] = 1.0f * wsf[0] + 1.0f * wsf[1];  // ALPHA=BETA=1
}

extern "C" void kernel_launch(void* const* d_in, const int* in_sizes, int n_in,
                              void* d_out, int out_size, void* d_ws, size_t ws_size,
                              hipStream_t stream) {
    const float*     feat    = (const float*)d_in[0];
    const long long* labels  = (const long long*)d_in[1];   // reference dtype int64
    const float*     centers = (const float*)d_in[2];
    const float*     W       = (const float*)d_in[3];
    const float*     bias    = (const float*)d_in[4];
    float* wsf = (float*)d_ws;
    float* out = (float*)d_out;
    unsigned short* whi = (unsigned short*)((char*)d_ws + WS_WHI_BYTES);
    unsigned short* wlo = (unsigned short*)((char*)d_ws + WS_WLO_BYTES);

    k_init<<<1, 32, 0, stream>>>(wsf);
    k_fisher_tlogit<<<B_ROWS / 8, 256, 0, stream>>>(feat, labels, centers, W, bias, wsf);
    if (ws_size >= (size_t)WS_NEEDED) {
        // fast path: one-time f32 -> bf16 hi/lo split of W, inner loop is pure load+WMMA
        k_convW<<<(C_CLS * F_DIM / 4 + 255) / 256, 256, 0, stream>>>(W, whi, wlo);
        k_logits_lse<true><<<B_ROWS / MB, 256, 0, stream>>>(feat, W, whi, wlo, bias, wsf);
    } else {
        // fallback: convert W fragments in-loop (more VALU, no extra workspace)
        k_logits_lse<false><<<B_ROWS / MB, 256, 0, stream>>>(feat, W, whi, wlo, bias, wsf);
    }
    k_final<<<1, 1, 0, stream>>>(wsf, out);
}
